// MicroAdder_20280835572161
// MI455X (gfx1250) — compile-verified, matrix operand order checked by reading, not run on hardware
//
#include <hip/hip_runtime.h>
#include <math.h>

#define B_TOT   65536
#define T_SEQ   34
#define TOTAL   (B_TOT * T_SEQ)     // 2,228,224 rows (fits int32)
#define VOCAB   14
#define DM      5
#define QK      5
#define POSD    3
#define FFNH    2
#define NB      7                   // batch elements per block
#define ROWS    (NB * T_SEQ)        // 238 (b,t) rows per block
#define TILES   ((ROWS + 15) / 16)  // 15 WMMA row-tiles
#define EPS     1e-5f
#define TWO_PI  6.28318530717958647692f

typedef __attribute__((ext_vector_type(2)))  float    v2f;
typedef __attribute__((ext_vector_type(8)))  float    v8f;
typedef __attribute__((ext_vector_type(16))) _Float16 v16h;

__global__ __launch_bounds__(256) void micro_fwd_kernel(
    const int* __restrict__ idx,
    const float* __restrict__ tok_A_p, const float* __restrict__ tok_start_p,
    const float* __restrict__ tok_stride_p,
    const float* __restrict__ sp_amp_p, const float* __restrict__ sp_phase_p,
    const float* __restrict__ sp_slope_p, const float* __restrict__ sp_offset_p,
    const float* __restrict__ norm_w,
    const float* __restrict__ q_w,
    const float* __restrict__ q_phase_p,
    const float* __restrict__ out_A, const float* __restrict__ out_B,
    const float* __restrict__ fc1_w, const float* __restrict__ fc2_w,
    const float* __restrict__ head_w,
    float* __restrict__ out)
{
    __shared__ float s_k[T_SEQ][8];                       // k vectors (pos-only)
    __shared__ float s_attn[T_SEQ][36];                   // softmaxed attention rows
    __shared__ float s_h[ROWS][6];                        // rmsnorm(x) values
    __shared__ float s_p[TILES * 16][2];                  // head-plane projections
    __shared__ __align__(16) float s_logits[TILES * 16 * VOCAB]; // 240*14 staged logits

    const int tid  = threadIdx.x;
    const int lane = tid & 31;
    const int wave = tid >> 5;

    const float tokA = tok_A_p[0], tok0 = tok_start_p[0], tokS = tok_stride_p[0];
    const float spA  = sp_amp_p[0], spPh = sp_phase_p[0];
    const float spSl = sp_slope_p[0], spOf = sp_offset_p[0];

    const int row_base = (int)blockIdx.x * ROWS;          // global flat row of r_loc==0

    // gfx1250 global_prefetch_b8: warm this block's idx slab
    if (tid == 0) __builtin_prefetch((const void*)(idx + row_base), 0, 1);

    const int r_loc = tid;                                // block-local row id
    const int bl    = r_loc / T_SEQ;                      // local batch element
    const int t     = r_loc - bl * T_SEQ;                 // sequence position
    const int grow  = row_base + r_loc;                   // global flat row b*T + t
    const bool valid = (r_loc < ROWS) && (grow < TOTAL);

    // ---- phase 1: k = pos @ q_w^T (threads 0..T-1) ----
    if (tid < T_SEQ) {
        float tt = (float)tid;
        float th = TWO_PI * tt * 0.1f + spPh;
        float p0 = spA * __cosf(th);
        float p1 = spA * __sinf(th);
        float p2 = spSl * tt + spOf;
        #pragma unroll
        for (int j = 0; j < QK; ++j)
            s_k[tid][j] = q_w[j * POSD + 0] * p0 + q_w[j * POSD + 1] * p1
                        + q_w[j * POSD + 2] * p2;
    }
    __syncthreads();

    // ---- phase 2a: causal softmax attention rows (threads 0..T-1) ----
    if (tid < T_SEQ) {
        float c = __cosf(q_phase_p[0]), s = __sinf(q_phase_p[0]);
        float k0 = s_k[tid][0], k1 = s_k[tid][1];
        float qv[QK];
        qv[0] = c * k0 - s * k1;
        qv[1] = s * k0 + c * k1;
        qv[2] = s_k[tid][2]; qv[3] = s_k[tid][3]; qv[4] = s_k[tid][4];
        const float scale = 0.44721359549995793928f;      // 1/sqrt(QK)
        float m = -1e30f;
        for (int ss = 0; ss <= tid; ++ss) {
            float d = 0.f;
            #pragma unroll
            for (int j = 0; j < QK; ++j) d += qv[j] * s_k[ss][j];
            d *= scale;
            s_attn[tid][ss] = d;
            m = fmaxf(m, d);
        }
        float sum = 0.f;
        for (int ss = 0; ss <= tid; ++ss) {
            float e = __expf(s_attn[tid][ss] - m);
            s_attn[tid][ss] = e;
            sum += e;
        }
        float inv = 1.f / sum;
        for (int ss = 0; ss <= tid; ++ss) s_attn[tid][ss] *= inv;
    }

    // ---- phase 2b: embedding + pre-norm values (all valid rows) ----
    float x[DM];
    float nw[DM];
    if (valid) {
        #pragma unroll
        for (int d = 0; d < DM; ++d) nw[d] = norm_w[d];
        int tok = idx[grow];
        float ang = tok0 + (float)tok * tokS;
        x[0] = tokA * __cosf(ang);
        x[1] = tokA * __sinf(ang);
        float tt = (float)t;
        float th = TWO_PI * tt * 0.1f + spPh;
        x[2] = spA * __cosf(th);
        x[3] = spA * __sinf(th);
        x[4] = spSl * tt + spOf;
        float ss2 = 0.f;
        #pragma unroll
        for (int d = 0; d < DM; ++d) ss2 += x[d] * x[d];
        float inv = rsqrtf(ss2 * (1.f / DM) + EPS);
        #pragma unroll
        for (int d = 0; d < DM; ++d) s_h[r_loc][d] = x[d] * inv * nw[d];
    }
    __syncthreads();

    // ---- phase 3: attention mix + residual + FFN + head projection ----
    float p0 = 0.f, p1 = 0.f;
    if (valid) {
        float acc[DM] = {0.f, 0.f, 0.f, 0.f, 0.f};
        int base = bl * T_SEQ;
        for (int ss = 0; ss <= t; ++ss) {
            float w = s_attn[t][ss];
            #pragma unroll
            for (int d = 0; d < DM; ++d) acc[d] += w * s_h[base + ss][d];
        }
        // rank-1 out_proj + residual
        float g = 0.f;
        #pragma unroll
        for (int d = 0; d < DM; ++d) g += acc[d] * out_A[d];
        #pragma unroll
        for (int d = 0; d < DM; ++d) x[d] += g * out_B[d];
        // FFN (pre-norm, 2 hidden units, relu)
        float ss2 = 0.f;
        #pragma unroll
        for (int d = 0; d < DM; ++d) ss2 += x[d] * x[d];
        float inv = rsqrtf(ss2 * (1.f / DM) + EPS);
        float h2[DM];
        #pragma unroll
        for (int d = 0; d < DM; ++d) h2[d] = x[d] * inv * nw[d];
        #pragma unroll
        for (int j = 0; j < FFNH; ++j) {
            float u = 0.f;
            #pragma unroll
            for (int d = 0; d < DM; ++d) u += h2[d] * fc1_w[j * DM + d];
            u = fmaxf(u, 0.f);
            #pragma unroll
            for (int d = 0; d < DM; ++d) x[d] += u * fc2_w[d * FFNH + j];
        }
        // head projection to token plane
        ss2 = 0.f;
        #pragma unroll
        for (int d = 0; d < DM; ++d) ss2 += x[d] * x[d];
        inv = rsqrtf(ss2 * (1.f / DM) + EPS);
        float hf[DM];
        #pragma unroll
        for (int d = 0; d < DM; ++d) hf[d] = x[d] * inv * nw[d];
        #pragma unroll
        for (int d = 0; d < DM; ++d) {
            p0 += hf[d] * head_w[0 * DM + d];
            p1 += hf[d] * head_w[1 * DM + d];
        }
    }
    if (tid < TILES * 16) { s_p[tid][0] = p0; s_p[tid][1] = p1; }
    __syncthreads();

    // ---- phase 4: logits = (p @ tok_emb^T) via WMMA; stage D tiles into LDS ----
    float b0 = 0.f, b1 = 0.f;
    if (lane < VOCAB) {
        float ang = tok0 + (float)lane * tokS;
        b0 = tokA * __cosf(ang);
        b1 = tokA * __sinf(ang);
    }
    int ti_end = wave * 2 + 2;
    if (ti_end > TILES) ti_end = TILES;
    for (int ti = wave * 2; ti < ti_end; ++ti) {          // wave-uniform bounds
        // unconditional padded-LDS read + lane select (no exec save/restore)
        int prow = ti * 16 + (lane & 15);
        float q0 = s_p[prow][0];
        float q1 = s_p[prow][1];
        v2f a;  a.x = (lane < 16) ? q0 : 0.f;             // A: K=0,1 on lanes 0-15
        a.y      = (lane < 16) ? q1 : 0.f;                // lanes 16-31 = K=2,3 pad
        v2f bm; bm.x = b0; bm.y = b1;                     // cols>=14 already zero
        v8f cz = {};
        v8f dres;
#if __has_builtin(__builtin_amdgcn_wmma_f32_16x16x4_f32)
        dres = __builtin_amdgcn_wmma_f32_16x16x4_f32(
            false, a, false, bm, (short)0, cz, false, false);
#else
        v16h ah = {};
        v16h bh = {};
        if (lane < 16) {
            ah[0] = (_Float16)a.x; ah[1] = (_Float16)a.y;
            bh[0] = (_Float16)b0;  bh[1] = (_Float16)b1;
        }
        dres = __builtin_amdgcn_wmma_f32_16x16x32_f16(
            false, ah, false, bh, (short)0, cz, false, false);
#endif
        int col   = lane & 15;
        int rbase = ti * 16 + ((lane >= 16) ? 8 : 0);     // D layout: vgpr r = M=r / r+8
        if (col < VOCAB) {
            #pragma unroll
            for (int r = 0; r < 8; ++r)
                s_logits[(rbase + r) * VOCAB + col] = dres[r];
        }
    }
    __syncthreads();

    // ---- phase 5: dense coalesced writeback, LDS -> HBM ----
    // Block owns out[row_base*14 .. +vrows*14): vrows is even => bytes % 16 == 0,
    // and row_base*14*4 % 16 == 0, so pure b128 chunks, no scalar tail.
    int rows_left = TOTAL - row_base;
    int vrows     = rows_left < ROWS ? rows_left : ROWS;
    int nchunks   = (vrows * VOCAB) >> 2;                 // float4 chunks (833 or 238)
    unsigned long long gbase = (unsigned long long)(out + (size_t)row_base * VOCAB);
    for (int i = tid; i < nchunks; i += 256) {
        unsigned goff  = (unsigned)(i * 16);              // byte offset from gbase
        unsigned laddr = (unsigned)(size_t)(&s_logits[i * 4]); // LDS byte address
        // gfx1250 async DMA: LDS -> global, 16B per lane, tracked on ASYNCcnt
        asm volatile("global_store_async_from_lds_b128 %0, %1, %2"
                     :: "v"(goff), "v"(laddr), "s"(gbase) : "memory");
    }
    asm volatile("s_wait_asynccnt 0" ::: "memory");
}

extern "C" void kernel_launch(void* const* d_in, const int* in_sizes, int n_in,
                              void* d_out, int out_size, void* d_ws, size_t ws_size,
                              hipStream_t stream) {
    const int*   idx        = (const int*)  d_in[0];
    const float* tok_A      = (const float*)d_in[1];
    const float* tok_start  = (const float*)d_in[2];
    const float* tok_stride = (const float*)d_in[3];
    const float* sp_amp     = (const float*)d_in[4];
    const float* sp_phase   = (const float*)d_in[5];
    const float* sp_slope   = (const float*)d_in[6];
    const float* sp_offset  = (const float*)d_in[7];
    const float* norm_w     = (const float*)d_in[8];
    const float* q_w        = (const float*)d_in[9];
    const float* q_phase    = (const float*)d_in[10];
    const float* out_A      = (const float*)d_in[11];
    const float* out_B      = (const float*)d_in[12];
    const float* fc1_w      = (const float*)d_in[13];
    const float* fc2_w      = (const float*)d_in[14];
    const float* head_w     = (const float*)d_in[15];
    float*       out        = (float*)d_out;

    const int nblocks = (TOTAL + ROWS - 1) / ROWS;        // 9363

    micro_fwd_kernel<<<nblocks, 256, 0, stream>>>(
        idx, tok_A, tok_start, tok_stride, sp_amp, sp_phase, sp_slope, sp_offset,
        norm_w, q_w, q_phase, out_A, out_B, fc1_w, fc2_w, head_w, out);
}
